// SimpleRNN_88390426952197
// MI455X (gfx1250) — matrix operand order repaired and checked by reading, not compiled, not run on hardware
//
#include <hip/hip_runtime.h>
#include <cmath>

// ---------------------------------------------------------------------------
// SimpleRNN on MI455X (gfx1250, wave32):
//   Phase 1: xw = x @ W_x^T + b_x        (bf16 WMMA, f32 accumulate)
//   Phase 2: h_t = tanh(h@W_h^T + xw_t)  (persistent kernel, W_h + h in LDS,
//            TDM tensor_load_to_lds staging, per-step device barrier)
// LDS layout uses a padded pitch: each 1024-elem bf16 row is stored as
// 2*(1024B data + 8B pad) = 2064 B = 516 dwords  ->  4-bank lane separation
// (conflict-free b128 fragment reads). TDM reproduces this with
// pad_interval=256 dwords, pad_amount=2 dwords.
// ---------------------------------------------------------------------------

typedef __attribute__((ext_vector_type(16))) __bf16 v16bf;
typedef __attribute__((ext_vector_type(8)))  __bf16 v8bf;
typedef __attribute__((ext_vector_type(4)))  __bf16 v4bf;
typedef __attribute__((ext_vector_type(8)))  float  v8f;
typedef __attribute__((ext_vector_type(4)))  unsigned int u32x4;
typedef __attribute__((ext_vector_type(8)))  int i32x8;
typedef __attribute__((ext_vector_type(4)))  int i32x4;

#define RNN_B    32
#define RNN_L    512
#define RNN_DIN  1024
#define RNN_DH   1024
#define RNN_NWG  16
#define COLS_PER_WG (RNN_DH / RNN_NWG)     // 64 columns per workgroup
#define PITCH_E  1032                      // padded LDS row pitch in elements
#define LDSH_BASE (COLS_PER_WG * PITCH_E)  // h-buffer base (after W_h slice)

// Padded LDS element offset for (row, k): +4 elements once k crosses 512.
__device__ __forceinline__ int lds_idx(int row, int k) {
  return row * PITCH_E + k + ((k >= 512) ? 4 : 0);
}

// ---- WMMA helpers ---------------------------------------------------------

__device__ __forceinline__ v8f wmma_bf16(v16bf a, v16bf b, v8f c) {
  return __builtin_amdgcn_wmma_f32_16x16x32_bf16(
      false, a, false, b, (short)0, c, false, false);
}

// A fragment (16x32): elems 0..7 = K+{0..7}, elems 8..15 = K+{16..23},
// with K base = k0 + 8*half (lane r holds row m0+r).
__device__ __forceinline__ v16bf frag_a(const __bf16* p) {
  v8bf lo = *(const v8bf*)(p);
  v8bf hi = *(const v8bf*)(p + 16);
  return __builtin_shufflevector(lo, hi, 0,1,2,3,4,5,6,7,8,9,10,11,12,13,14,15);
}

// B fragment (32x16): elems 0..7 = K+{0..7}, elems 8..15 = K+{16..23}, K = k0 + 8*half, column n0+r
// (a column of B is a contiguous row of W).
__device__ __forceinline__ v16bf frag_b(const __bf16* p) {
  v8bf lo = *(const v8bf*)(p);
  v8bf hi = *(const v8bf*)(p + 16);
  return __builtin_shufflevector(lo, hi, 0,1,2,3,4,5,6,7,8,9,10,11,12,13,14,15);
}

// ---- TDM: 2D tile (rows x 1024 bf16) global -> LDS with padded pitch ------
// D# per CDNA5 ISA: group0 = {count/lds/global/type},
// group1 = {data_size, pad cfg, dims, tile dims, stride}. This toolchain's
// builtin takes 6 args (g0, g1, g2, g3, extra, cpol); groups 2/3 + extra are
// zero-filled (2D tile: tile_dim2/3 = 0 -> unused).

#if __has_builtin(__builtin_amdgcn_tensor_load_to_lds) && \
    __has_builtin(__builtin_amdgcn_s_wait_tensorcnt)
#define RNN_HAS_TDM 1
__device__ __forceinline__ void tdm_load_rows(unsigned int lds_byte_addr,
                                              const __bf16* gsrc,
                                              unsigned int rows) {
  unsigned long long ga = (unsigned long long)(uintptr_t)gsrc;
  u32x4 g0;
  g0[0] = 1u;                                    // count=1, user descriptor
  g0[1] = lds_byte_addr;                         // LDS dest (bytes)
  g0[2] = (unsigned int)ga;                      // global addr [31:0]
  g0[3] = (unsigned int)((ga >> 32) & 0x01FFFFFFu) | (2u << 30); // addr[56:32]|type=2
  i32x8 g1;
  // data_size=2B(code1) | pad_enable | pad_interval=256dw(code7) |
  // pad_amount=2dw(code1)
  g1[0] = (int)((1u << 16) | (1u << 20) | (7u << 22) | (1u << 25));
  g1[1] = (int)(1024u << 16);                    // tensor_dim0[15:0] = 1024
  g1[2] = (int)(rows << 16);                     // tensor_dim1[15:0] = rows
  g1[3] = (int)(1024u << 16);                    // tile_dim0 = 1024
  g1[4] = (int)rows;                             // tile_dim1 = rows, tile_dim2=0
  g1[5] = 1024;                                  // tensor_dim0_stride[31:0]
  g1[6] = 0;
  g1[7] = 0;
  i32x4 z4 = {0, 0, 0, 0};
  i32x8 z8 = {0, 0, 0, 0, 0, 0, 0, 0};
  __builtin_amdgcn_tensor_load_to_lds(g0, g1, z4, z4, z8, 0);
}
#else
#define RNN_HAS_TDM 0
#endif

// Manual padded staging fallback (v8bf -> ds_store_b128).
__device__ __forceinline__ void stage_rows_manual(__bf16* lds_dst_base,
                                                  const __bf16* gsrc,
                                                  int rows, int tid, int nthr) {
  for (int idx = tid; idx < rows * 128; idx += nthr) {
    const int row = idx >> 7;
    const int k   = (idx & 127) * 8;           // multiple of 8, never crosses 512
    *(v8bf*)(&lds_dst_base[lds_idx(row, k)]) =
        *(const v8bf*)(&gsrc[(size_t)row * 1024 + k]);
  }
}

// ---- Phase 0: f32 -> bf16 (x4 vectorized) ---------------------------------

__global__ void rnn_cvt_bf16(const float* __restrict__ s,
                             __bf16* __restrict__ d, int n4) {
  for (int i = blockIdx.x * blockDim.x + threadIdx.x; i < n4;
       i += gridDim.x * blockDim.x) {
    const float4 f = ((const float4*)s)[i];
    v4bf o; o[0] = (__bf16)f.x; o[1] = (__bf16)f.y;
    o[2] = (__bf16)f.z; o[3] = (__bf16)f.w;
    ((v4bf*)d)[i] = o;
  }
}

// ---- Phase 1: xw[m,n] = sum_k x[m,k]*W_x[n,k] + b_x[n] --------------------
// One wave = 16(M) x 64(N) tile; all 4 B fragments issued before the 4 WMMAs
// so the waits become loadcnt<=3..0 instead of 0 (overlap loads with WMMA).

__global__ void rnn_xw_gemm(const __bf16* __restrict__ xbf,
                            const __bf16* __restrict__ wxbf,
                            const float*  __restrict__ bx,
                            float* __restrict__ xw) {
  const int lane = threadIdx.x & 31;
  const int r    = lane & 15;
  const int half = lane >> 4;
  const int wave = blockIdx.x * (blockDim.x >> 5) + (threadIdx.x >> 5);
  const int m0 = (wave >> 4) * 16;
  const int nb = (wave & 15) * 64;

  v8f acc[4];
  #pragma unroll
  for (int s = 0; s < 4; ++s) acc[s] = (v8f)(0.0f);

  const __bf16* ap = xbf + (size_t)(m0 + r) * RNN_DIN + half * 8;
  const __bf16* bp = wxbf + (size_t)(nb + r) * RNN_DIN + half * 8;

  for (int k0 = 0; k0 < RNN_DIN; k0 += 32) {
    v16bf a = frag_a(ap + k0);
    v16bf bf[4];
    #pragma unroll
    for (int s = 0; s < 4; ++s)
      bf[s] = frag_b(bp + (size_t)s * 16 * RNN_DIN + k0);
    #pragma unroll
    for (int s = 0; s < 4; ++s)
      acc[s] = wmma_bf16(a, bf[s], acc[s]);
  }

  #pragma unroll
  for (int s = 0; s < 4; ++s) {
    const int n = nb + s * 16 + r;
    const float bias = bx[n];
    #pragma unroll
    for (int i = 0; i < 8; ++i) {
      const int m = m0 + i + half * 8;
      xw[(size_t)m * RNN_DH + n] = acc[s][i] + bias;
    }
  }
}

// ---- Phase 2 init ---------------------------------------------------------

__global__ void rnn_init(__bf16* __restrict__ h0, int* __restrict__ counter) {
  const int i = blockIdx.x * blockDim.x + threadIdx.x;
  if (i < RNN_B * RNN_DH) h0[i] = (__bf16)0.0f;
  if (i == 0) *counter = 0;
}

// ---- Phase 2: persistent recurrence ---------------------------------------
// 16 blocks x 256 threads (8 waves). Block wg owns columns [wg*64, wg*64+64).
// LDS: W_h slice (64 rows, padded pitch) + current h (32 rows, padded pitch).

__global__ void rnn_steps(const __bf16* __restrict__ whbf,
                          const float*  __restrict__ xw,
                          __bf16* __restrict__ hbuf0,
                          __bf16* __restrict__ hbuf1,
                          float* __restrict__ out,
                          int* __restrict__ counter) {
  extern __shared__ __bf16 smem[];     // [64*PITCH_E] W_h  + [32*PITCH_E] h
  const int wg   = blockIdx.x;
  const int tid  = threadIdx.x;
  const bool tdm_wave = (tid >> 5) == 0;

  // ---- Stage W_h slice once (TDM if available) ----
  const __bf16* whsrc = whbf + (size_t)wg * COLS_PER_WG * RNN_DH;
#if RNN_HAS_TDM
  if (tdm_wave) {
    tdm_load_rows((unsigned int)(uintptr_t)&smem[0], whsrc, COLS_PER_WG);
    __builtin_amdgcn_s_wait_tensorcnt(0);
  }
#else
  stage_rows_manual(&smem[0], whsrc, COLS_PER_WG, tid, blockDim.x);
#endif
  __syncthreads();

  const int lane = tid & 31;
  const int r    = lane & 15;
  const int half = lane >> 4;
  const int wave = tid >> 5;
  const int m0   = (wave & 1) * 16;         // batch tile (0 or 16)
  const int nl0  = (wave >> 1) * 16;        // local column tile
  const int n0   = wg * COLS_PER_WG + nl0;  // global column tile
  const int n    = n0 + r;

  const __bf16* aBase = &smem[LDSH_BASE + (m0 + r) * PITCH_E + half * 8];
  const __bf16* bBase = &smem[(nl0 + r) * PITCH_E + half * 8];

  __bf16* bufs[2] = {hbuf0, hbuf1};

  for (int t = 0; t < RNN_L; ++t) {
    const __bf16* hcur  = bufs[t & 1];
    __bf16*       hnext = bufs[(t + 1) & 1];

    // Prefetch the epilogue xw values (read-once -> non-temporal); these
    // global loads overlap the staging + entire K loop.
    float xv[8];
    #pragma unroll
    for (int i = 0; i < 8; ++i) {
      const int batch = m0 + i + half * 8;
      xv[i] = __builtin_nontemporal_load(
          &xw[((size_t)batch * RNN_L + t) * RNN_DH + n]);
    }

    // ---- Stage current h into LDS ----
#if RNN_HAS_TDM
    if (tdm_wave) {
      tdm_load_rows((unsigned int)(uintptr_t)&smem[LDSH_BASE], hcur, RNN_B);
      __builtin_amdgcn_s_wait_tensorcnt(0);
    }
#else
    stage_rows_manual(&smem[LDSH_BASE], hcur, RNN_B, tid, blockDim.x);
#endif
    __syncthreads();

    // ---- K loop, software-pipelined one fragment ahead (all ds_load) ----
    v8f acc = (v8f)(0.0f);
    v16bf a = frag_a(aBase);          // k0 = 0 (side offset 0)
    v16bf b = frag_b(bBase);
    #pragma unroll 4
    for (int k0 = 0; k0 < RNN_DH - 32; k0 += 32) {
      const int kn = k0 + 32;
      const int ko = kn + ((kn >= 512) ? 4 : 0);
      v16bf an = frag_a(aBase + ko);
      v16bf bn = frag_b(bBase + ko);
      acc = wmma_bf16(a, b, acc);
      a = an; b = bn;
    }
    acc = wmma_bf16(a, b, acc);

    // ---- Epilogue: tanh, stream out (NT), update h state ----
    #pragma unroll
    for (int i = 0; i < 8; ++i) {
      const int batch = m0 + i + half * 8;
      const float v = tanhf(acc[i] + xv[i]);
      __builtin_nontemporal_store(v,
          &out[((size_t)batch * RNN_L + t) * RNN_DH + n]);
      hnext[(size_t)batch * RNN_DH + n] = (__bf16)v;
    }

    // ---- Device-wide barrier: release, arrive, spin, acquire ----
    __builtin_amdgcn_fence(__ATOMIC_RELEASE, "agent");
    __syncthreads();
    if (tid == 0) {
      __hip_atomic_fetch_add(counter, 1, __ATOMIC_RELEASE,
                             __HIP_MEMORY_SCOPE_AGENT);
      const int target = RNN_NWG * (t + 1);
      while (__hip_atomic_load(counter, __ATOMIC_ACQUIRE,
                               __HIP_MEMORY_SCOPE_AGENT) < target)
        __builtin_amdgcn_s_sleep(1);
    }
    __syncthreads();
    __builtin_amdgcn_fence(__ATOMIC_ACQUIRE, "agent");
  }
}

// ---- Host-side orchestration ----------------------------------------------

extern "C" void kernel_launch(void* const* d_in, const int* in_sizes, int n_in,
                              void* d_out, int out_size, void* d_ws,
                              size_t ws_size, hipStream_t stream) {
  // setup_inputs order: x, W_h, W_x, b_x (all f32)
  const float* x  = (const float*)d_in[0];
  const float* Wh = (const float*)d_in[1];
  const float* Wx = (const float*)d_in[2];
  const float* bx = (const float*)d_in[3];
  float* out = (float*)d_out;

  const size_t nX  = (size_t)RNN_B * RNN_L * RNN_DIN;
  const size_t nWh = (size_t)RNN_DH * RNN_DH;
  const size_t nWx = (size_t)RNN_DH * RNN_DIN;
  const size_t nXW = (size_t)RNN_B * RNN_L * RNN_DH;
  const size_t nH  = (size_t)RNN_B * RNN_DH;

  char* ws = (char*)d_ws;
  __bf16* whbf = (__bf16*)(ws);
  __bf16* wxbf = (__bf16*)(ws + nWh * 2);
  __bf16* xbf  = (__bf16*)(ws + (nWh + nWx) * 2);
  float*  xw   = (float*)(ws + (nWh + nWx + nX) * 2);
  __bf16* h0   = (__bf16*)(ws + (nWh + nWx + nX) * 2 + nXW * 4);
  __bf16* h1   = h0 + nH;
  int*    cnt  = (int*)(h1 + nH);

  // Phase 0: f32 -> bf16
  rnn_cvt_bf16<<<2048, 256, 0, stream>>>(x,  xbf,  (int)(nX  / 4));
  rnn_cvt_bf16<<<512,  256, 0, stream>>>(Wh, whbf, (int)(nWh / 4));
  rnn_cvt_bf16<<<512,  256, 0, stream>>>(Wx, wxbf, (int)(nWx / 4));

  // Phase 1: 1024 M-tiles x 16 N-groups = 16384 waves -> 2048 blocks x 8 waves
  rnn_xw_gemm<<<2048, 256, 0, stream>>>(xbf, wxbf, bx, xw);

  // Phase 2: init + persistent recurrence (dynamic LDS = 96 padded rows)
  rnn_init<<<(int)((nH + 255) / 256), 256, 0, stream>>>(h0, cnt);
  const int lds_bytes = (COLS_PER_WG + RNN_B) * PITCH_E * (int)sizeof(__bf16);
  rnn_steps<<<RNN_NWG, 256, lds_bytes, stream>>>(whbf, xw, h0, h1, out, cnt);
}
